// GCN_Graph_59519656788288
// MI455X (gfx1250) — matrix operand order repaired and checked by reading, not compiled, not run on hardware
//
#include <hip/hip_runtime.h>
#include <hip/hip_bf16.h>

// Problem constants (match reference)
#define NN 100000   // nodes
#define NE 200000   // edges
#define NF 56       // atom features
#define H  256      // hidden
#define NG 2048     // graphs
#define BN_EPS 1e-5f

typedef __attribute__((ext_vector_type(16))) __bf16 v16bf;
typedef __attribute__((ext_vector_type(8)))  float  v8f;
typedef __attribute__((ext_vector_type(4)))  int    v4i;

#if defined(__HIP_DEVICE_COMPILE__) && __has_builtin(__builtin_amdgcn_global_load_async_to_lds_b128)
#define USE_ASYNC_LDS 1
typedef __attribute__((address_space(1))) v4i* gv4i_p;   // global int4*
typedef __attribute__((address_space(3))) v4i* lv4i_p;   // LDS int4*
#endif

static __device__ __forceinline__ unsigned short f2bf(float f) {
    unsigned int u = __builtin_bit_cast(unsigned int, f);
    u += 0x7FFFu + ((u >> 16) & 1u);   // round-to-nearest-even
    return (unsigned short)(u >> 16);
}

// ---------------- AtomEncoder: h[n][j] = sum_f emb[f, x[n,f], j]  (-> bf16) ----
__global__ void atom_encoder_kernel(const int* __restrict__ x,
                                    const float* __restrict__ emb,
                                    unsigned short* __restrict__ hbf) {
    const int n = blockIdx.x;
    const int j = threadIdx.x;
    float acc = 0.0f;
    #pragma unroll 8
    for (int f = 0; f < NF; ++f) {
        const int idx = x[n * NF + f];           // 0 or 1
        acc += emb[(f * 2 + idx) * H + j];
    }
    hbf[n * H + j] = f2bf(acc);
}

// ------- W -> bf16, TRANSPOSED: wbt[layer][n][k] = bf16(W[layer][k][n]) -------
__global__ void convert_wt_kernel(const float* __restrict__ Ws,
                                  unsigned short* __restrict__ wbt) {
    const int i = blockIdx.x * blockDim.x + threadIdx.x;
    if (i >= 3 * H * H) return;
    const int layer = i / (H * H);
    const int rem   = i - layer * H * H;
    const int n     = rem / H;                   // output row (column of W)
    const int k     = rem - n * H;               // output col (row of W)
    wbt[i] = f2bf(Ws[layer * H * H + k * H + n]);
}

// ---------------- degree with self loops --------------------------------------
__global__ void deg_init_kernel(float* __restrict__ deg) {
    const int i = blockIdx.x * blockDim.x + threadIdx.x;
    if (i < NN) deg[i] = 1.0f;                   // self loop contributes 1
}
__global__ void deg_accum_kernel(const int* __restrict__ ei, float* __restrict__ deg) {
    const int e = blockIdx.x * blockDim.x + threadIdx.x;
    if (e < NE) atomicAdd(&deg[ei[NE + e]], 1.0f);   // dst row of edge_index
}
__global__ void deg_finalize_kernel(float* __restrict__ deg) {
    const int i = blockIdx.x * blockDim.x + threadIdx.x;
    if (i < NN) deg[i] = rsqrtf(deg[i]);         // deg >= 1 always
}

// ---- WMMA GEMM: Out[M,256] = A_bf16[M,256] @ W[256,256]  (W given as WT) -----
// Block = 8 waves = 8 M-tiles x 128-col strip. The 64KB B strip is shared by
// all 8 waves -> staged once into LDS (async copy when available), then each
// WMMA is fed by ds_load_b128 (padded rows: bank-conflict-free).
#define NT 8                 // N-tiles per wave
#define BSTRIDE 264          // padded LDS row stride in ushorts (528B -> bank+4/row)
__global__ void gemm_bf16_kernel(const unsigned short* __restrict__ A,
                                 const unsigned short* __restrict__ WT,
                                 float* __restrict__ Out, int mtiles) {
    __shared__ __align__(16) unsigned short Bs[NT * 16 * BSTRIDE];   // ~66KB

    const int tid  = threadIdx.x;
    const int lane = tid & 31;
    const int wave = tid >> 5;
    const int n0   = blockIdx.y * (NT * 16);     // 128-column strip

    // ---- cooperative stage: 128 rows x 256 k of WT -> LDS (16B chunks) ----
    {
        const unsigned short* gsrc = WT + (size_t)n0 * H;
        #pragma unroll
        for (int it = 0; it < 16; ++it) {
            const int id  = it * 256 + tid;      // 4096 chunks of 16B
            const int row = id >> 5;
            const int c16 = id & 31;
#ifdef USE_ASYNC_LDS
            __builtin_amdgcn_global_load_async_to_lds_b128(
                (gv4i_p)(gsrc + (size_t)row * H + c16 * 8),
                (lv4i_p)(Bs + row * BSTRIDE + c16 * 8), 0, 0);
#else
            *(float4*)(Bs + row * BSTRIDE + c16 * 8) =
                *(const float4*)(gsrc + (size_t)row * H + c16 * 8);
#endif
        }
#ifdef USE_ASYNC_LDS
#if __has_builtin(__builtin_amdgcn_s_wait_asynccnt)
        __builtin_amdgcn_s_wait_asynccnt(0);
#else
        asm volatile("s_wait_asynccnt 0" ::: "memory");
#endif
#endif
    }
    __syncthreads();

    const int tileM = blockIdx.x * 8 + wave;     // wave-uniform
    if (tileM < mtiles) {                        // uniform branch: EXEC all-1s inside
        const int hb = lane >> 4;                // lane half (0/1)
        const int r  = lane & 15;
        const int m0 = tileM * 16;

        union Frag { unsigned int u[8]; v16bf v; };
        Frag a, b;
        v8f c[NT];
        #pragma unroll
        for (int t = 0; t < NT; ++t) c[t] = (v8f){};

        const unsigned short* arow = A + (size_t)(m0 + r) * H;
        // B fragment: lanes 0-15 hold K=0..15 (VGPR v -> K=2v,2v+1),
        // lanes 16-31 hold K=16..31; contiguous per lane -> ds_load_b128 x2.
        const unsigned short* blbase = Bs + r * BSTRIDE + 16 * hb;

        for (int kc = 0; kc < H; kc += 32) {
            #pragma unroll
            for (int v = 0; v < 8; ++v) {
                // A fragment, contiguous k order as for B: VGPR v holds K = 2*v + 16*half (+1)
                const int koff = 2 * v + 16 * hb;
                a.u[v] = *(const unsigned int*)(arow + kc + koff);   // 2x b128 total
            }
            #pragma unroll
            for (int t = 0; t < NT; ++t) {
                const unsigned short* bl = blbase + t * 16 * BSTRIDE + kc;
                #pragma unroll
                for (int v = 0; v < 8; ++v)
                    b.u[v] = *(const unsigned int*)(bl + 2 * v);     // LDS, contiguous
                c[t] = __builtin_amdgcn_wmma_f32_16x16x32_bf16(false, a.v, false, b.v,
                                                               (short)0, c[t], false, false);
            }
        }

        // C/D layout: VGPR i -> M = i + 8*half, N = lane&15
        #pragma unroll
        for (int t = 0; t < NT; ++t) {
            float* orow = Out + (size_t)(m0 + 8 * hb) * H + n0 + t * 16 + r;
            #pragma unroll
            for (int i = 0; i < 8; ++i) orow[(size_t)i * H] = c[t][i];
        }
    }
}

// ---------------- agg init with bias (matches segment_sum(msg) + b) -----------
__global__ void agg_init_kernel(const float* __restrict__ bias, float* __restrict__ agg) {
    agg[(size_t)blockIdx.x * H + threadIdx.x] = bias[threadIdx.x];
}

// ---------------- edge + self-loop scatter: agg[d] += hw[s] * norm ------------
__global__ void scatter_kernel(const int* __restrict__ ei,
                               const float* __restrict__ dinv,
                               const float* __restrict__ hw,
                               float* __restrict__ agg) {
    const int e = blockIdx.x;
    const int j = threadIdx.x;
    int s, d; float nrm;
    if (e < NE) {
        s = ei[e];
        d = ei[NE + e];
        nrm = dinv[s] * dinv[d];
    } else {                                     // self loop
        s = d = e - NE;
        const float di = dinv[s];
        nrm = di * di;
    }
    atomicAdd(&agg[(size_t)d * H + j], hw[(size_t)s * H + j] * nrm);
}

// ---------------- BatchNorm statistics (column sums, coalesced) ---------------
__global__ void zero_kernel(float* __restrict__ p, int n) {
    const int i = blockIdx.x * blockDim.x + threadIdx.x;
    if (i < n) p[i] = 0.0f;
}
__global__ void bn_stats_kernel(const float* __restrict__ agg, float* __restrict__ sums) {
    const int j = threadIdx.x;                   // column
    const int rows_per_block = (NN + gridDim.x - 1) / gridDim.x;
    const int r0 = blockIdx.x * rows_per_block;
    const int r1 = min(r0 + rows_per_block, NN);
    float s = 0.0f, s2 = 0.0f;
    for (int r = r0; r < r1; ++r) {
        const float v = agg[(size_t)r * H + j];  // threads read consecutive addrs
        s += v; s2 += v * v;
    }
    atomicAdd(&sums[j], s);
    atomicAdd(&sums[H + j], s2);
}

// ---------------- BN apply + ReLU -> bf16 for next layer ----------------------
__global__ void bn_relu_kernel(const float* __restrict__ agg,
                               const float* __restrict__ sums,
                               const float* __restrict__ gamma,
                               const float* __restrict__ beta,
                               unsigned short* __restrict__ hbf) {
    const int n = blockIdx.x;
    const int j = threadIdx.x;
    const float inv_n = 1.0f / (float)NN;
    const float mu  = sums[j] * inv_n;
    const float var = sums[H + j] * inv_n - mu * mu;   // ddof=0 (jnp.var default)
    const float is  = rsqrtf(var + BN_EPS);
    float y = (agg[(size_t)n * H + j] - mu) * is * gamma[j] + beta[j];
    y = fmaxf(y, 0.0f);
    hbf[(size_t)n * H + j] = f2bf(y);
}

// ---------------- global mean pool --------------------------------------------
__global__ void pool_accum_kernel(const float* __restrict__ hfinal,
                                  const int* __restrict__ batch,
                                  float* __restrict__ pooled,
                                  float* __restrict__ counts) {
    const int n = blockIdx.x;
    const int j = threadIdx.x;
    const int g = batch[n];
    atomicAdd(&pooled[(size_t)g * H + j], hfinal[(size_t)n * H + j]);
    if (j == 0) atomicAdd(&counts[g], 1.0f);
}

// ---------------- final linear: out[g] = (pooled[g]/cnt) . lin_w + lin_b ------
__global__ void final_linear_kernel(const float* __restrict__ pooled,
                                    const float* __restrict__ counts,
                                    const float* __restrict__ lw,
                                    const float* __restrict__ lb,
                                    float* __restrict__ out) {
    __shared__ float sd[H];
    const int g = blockIdx.x;
    const int j = threadIdx.x;
    const float c = fmaxf(counts[g], 1.0f);
    sd[j] = pooled[(size_t)g * H + j] / c * lw[j];
    __syncthreads();
    #pragma unroll
    for (int s = H / 2; s > 0; s >>= 1) {
        if (j < s) sd[j] += sd[j + s];
        __syncthreads();
    }
    if (j == 0) out[g] = sd[0] + lb[0];
}

extern "C" void kernel_launch(void* const* d_in, const int* in_sizes, int n_in,
                              void* d_out, int out_size, void* d_ws, size_t ws_size,
                              hipStream_t stream) {
    (void)in_sizes; (void)n_in; (void)out_size; (void)ws_size;

    const int*   x      = (const int*)d_in[0];     // [NN, 56]
    const int*   ei     = (const int*)d_in[1];     // [2, NE]
    const int*   batch  = (const int*)d_in[2];     // [NN]
    const float* emb    = (const float*)d_in[3];   // [56, 2, H]
    const float* Ws     = (const float*)d_in[4];   // [3, H, H]
    const float* bs     = (const float*)d_in[5];   // [3, H]
    const float* gammas = (const float*)d_in[6];   // [2, H]
    const float* betas  = (const float*)d_in[7];   // [2, H]
    const float* lin_w  = (const float*)d_in[8];   // [H, 1]
    const float* lin_b  = (const float*)d_in[9];   // [1]
    float* out = (float*)d_out;                    // [NG]

    // Workspace carve-out (all offsets 256B aligned)
    char* ws = (char*)d_ws;
    size_t off = 0;
    auto carve = [&](size_t bytes) -> void* {
        void* p = ws + off;
        off = (off + bytes + 255) & ~(size_t)255;
        return p;
    };
    unsigned short* hbf    = (unsigned short*)carve((size_t)NN * H * 2); // bf16 activations
    float*          hw     = (float*)carve((size_t)NN * H * 4);          // GEMM output
    float*          agg    = (float*)carve((size_t)NN * H * 4);          // scatter target
    float*          dinv   = (float*)carve((size_t)NN * 4);
    unsigned short* wbt    = (unsigned short*)carve((size_t)3 * H * H * 2); // W^T in bf16
    float*          sums   = (float*)carve((size_t)2 * H * 4);           // mu/var partials
    float*          pooled = (float*)carve((size_t)NG * H * 4);
    float*          counts = (float*)carve((size_t)NG * 4);

    const int MT = NN / 16;                       // 6250 exact M tiles

    // 1. AtomEncoder -> bf16 activations
    atom_encoder_kernel<<<NN, H, 0, stream>>>(x, emb, hbf);

    // 2. Weights -> bf16 transposed (all three layers at once)
    convert_wt_kernel<<<(3 * H * H + 255) / 256, 256, 0, stream>>>(Ws, wbt);

    // 3. GCN normalization (deg with self loops -> dinv)
    deg_init_kernel<<<(NN + 255) / 256, 256, 0, stream>>>(dinv);
    deg_accum_kernel<<<(NE + 255) / 256, 256, 0, stream>>>(ei, dinv);
    deg_finalize_kernel<<<(NN + 255) / 256, 256, 0, stream>>>(dinv);

    // 4. Three GCN layers
    for (int layer = 0; layer < 3; ++layer) {
        dim3 ggrid((MT + 7) / 8, H / (NT * 16));  // (782, 2)
        gemm_bf16_kernel<<<ggrid, 256, 0, stream>>>(hbf, wbt + (size_t)layer * H * H, hw, MT);

        agg_init_kernel<<<NN, H, 0, stream>>>(bs + layer * H, agg);
        scatter_kernel<<<NE + NN, H, 0, stream>>>(ei, dinv, hw, agg);

        if (layer < 2) {
            zero_kernel<<<(2 * H + 255) / 256, 256, 0, stream>>>(sums, 2 * H);
            bn_stats_kernel<<<256, H, 0, stream>>>(agg, sums);
            bn_relu_kernel<<<NN, H, 0, stream>>>(agg, sums, gammas + layer * H,
                                                 betas + layer * H, hbf);
        }
    }

    // 5. Global mean pool + final linear
    zero_kernel<<<(NG * H + 255) / 256, 256, 0, stream>>>(pooled, NG * H);
    zero_kernel<<<(NG + 255) / 256, 256, 0, stream>>>(counts, NG);
    pool_accum_kernel<<<NN, H, 0, stream>>>(agg, batch, pooled, counts);
    final_linear_kernel<<<NG, H, 0, stream>>>(pooled, counts, lin_w, lin_b, out);
}
